// TextQuerySegDetector_47476568490614
// MI455X (gfx1250) — matrix operand order repaired and checked by reading, not compiled, not run on hardware
//
#include <hip/hip_runtime.h>

// ---------------------------------------------------------------------------
// Types
// ---------------------------------------------------------------------------
typedef _Float16 h16;
typedef __attribute__((ext_vector_type(16))) _Float16 v16h;
typedef __attribute__((ext_vector_type(8)))  _Float16 v8h;
typedef __attribute__((ext_vector_type(8)))  float    v8f;

// ---------------------------------------------------------------------------
// WMMA GEMM:  C[M,N] (f32) = A[M,K] (f16, row-major) x Bt[N,K]^T (f16, row-major)
// epilogue: C = relu?( acc*scaleM[m] + biasM[m] + biasN[n] )
// Block tile 64x128, K-step 32; 256 threads = 8 wave32 arranged 2(M) x 4(N),
// each wave a 32x32 C tile (4 x v_wmma_f32_16x16x32_f16 per K-step).
// Double-buffered LDS staging (reg-staged) + global_prefetch for tile k+2.
// ---------------------------------------------------------------------------
#define TBM 64
#define TBN 128
#define TBK 32
#define LDS_STRIDE 40  // 32 + 8 halfs pad -> 80B rows (16B aligned, bank spread)

__device__ __forceinline__ v16h frag_a(const h16* row) {
  // elements 0..7  : K = 8*half + i      (at row+0)
  // elements 8..15 : K = 16 + 8*half + i (at row+16)
  v8h lo = *(const v8h*)(row);
  v8h hi = *(const v8h*)(row + 16);
  v16h r;
#pragma unroll
  for (int i = 0; i < 8; ++i) { r[i] = lo[i]; r[i + 8] = hi[i]; }
  return r;
}
__device__ __forceinline__ v16h frag_b(const h16* row) {
  // elements 0..7 : K = 8*half + i (row+0); elements 8..15 : K = 16 + 8*half + i (row+16)
  v8h lo = *(const v8h*)(row);
  v8h hi = *(const v8h*)(row + 16);
  v16h r;
#pragma unroll
  for (int i = 0; i < 8; ++i) { r[i] = lo[i]; r[i + 8] = hi[i]; }
  return r;
}

__global__ __launch_bounds__(256) void gemm_nt(
    const h16* __restrict__ A, const h16* __restrict__ Bm, float* __restrict__ C,
    int M, int N, int K,
    const float* __restrict__ scaleM, const float* __restrict__ biasM,
    const float* __restrict__ biasN, int relu)
{
  __shared__ __align__(16) h16 As[2][TBM * LDS_STRIDE];
  __shared__ __align__(16) h16 Bs[2][TBN * LDS_STRIDE];

  const int tid  = threadIdx.x;
  const int lane = tid & 31;
  const int wave = tid >> 5;
  const int wm   = wave >> 2;   // 0..1
  const int wn   = wave & 3;    // 0..3
  const int mBlk = blockIdx.y * TBM;
  const int nBlk = blockIdx.x * TBN;
  const int half = lane >> 4;
  const int l16  = lane & 15;

  // staging coordinates: A tile 64x32 = 256 chunks of 8h (1/thread);
  //                      B tile 128x32 = 512 chunks (2/thread)
  const int ar  = tid >> 2;          // A row 0..63
  const int akc = (tid & 3) * 8;
  const int br0 = tid >> 2;          // B chunk 0 row (chunks 0..255)
  const int bk0 = (tid & 3) * 8;
  const int br1 = (tid + 256) >> 2;  // B chunk 1 row (chunks 256..511)
  const int bk1 = bk0;

  v8h aReg, bReg0, bReg1;
  auto g_load = [&](int k0) {
    aReg = {}; bReg0 = {}; bReg1 = {};
    int gm = mBlk + ar;
    if (gm < M && (k0 + akc) < K)
      aReg = *(const v8h*)(A + (size_t)gm * K + k0 + akc);
    int gn0 = nBlk + br0;
    if (gn0 < N && (k0 + bk0) < K)
      bReg0 = *(const v8h*)(Bm + (size_t)gn0 * K + k0 + bk0);
    int gn1 = nBlk + br1;
    if (gn1 < N && (k0 + bk1) < K)
      bReg1 = *(const v8h*)(Bm + (size_t)gn1 * K + k0 + bk1);
  };
  auto s_store = [&](int buf) {
    *(v8h*)(&As[buf][ar  * LDS_STRIDE + akc]) = aReg;
    *(v8h*)(&Bs[buf][br0 * LDS_STRIDE + bk0]) = bReg0;
    *(v8h*)(&Bs[buf][br1 * LDS_STRIDE + bk1]) = bReg1;
  };

  v8f acc00 = {}; v8f acc01 = {}; v8f acc10 = {}; v8f acc11 = {};

  const int nk = (K + TBK - 1) / TBK;
  g_load(0);
  s_store(0);
  __syncthreads();

  for (int kt = 0; kt < nk; ++kt) {
    const int cur = kt & 1;
    if (kt + 1 < nk) g_load((kt + 1) * TBK);               // overlap with WMMA
    if (kt + 2 < nk) {                                     // pull k+2 toward L2
      int kp = (kt + 2) * TBK;
      __builtin_prefetch(A  + (size_t)(mBlk + ar)  * K + kp + akc, 0, 3);
      __builtin_prefetch(Bm + (size_t)(nBlk + br0) * K + kp + bk0, 0, 3);
      __builtin_prefetch(Bm + (size_t)(nBlk + br1) * K + kp + bk1, 0, 3);
    }

    const h16* Ab = As[cur];
    const h16* Bb = Bs[cur];
    v16h a0 = frag_a(&Ab[(wm * 32 +      l16) * LDS_STRIDE + 8 * half]);
    v16h a1 = frag_a(&Ab[(wm * 32 + 16 + l16) * LDS_STRIDE + 8 * half]);
    v16h b0 = frag_b(&Bb[(wn * 32 +      l16) * LDS_STRIDE + 8 * half]);
    v16h b1 = frag_b(&Bb[(wn * 32 + 16 + l16) * LDS_STRIDE + 8 * half]);

    acc00 = __builtin_amdgcn_wmma_f32_16x16x32_f16(false, a0, false, b0, (short)0, acc00, false, false);
    acc01 = __builtin_amdgcn_wmma_f32_16x16x32_f16(false, a0, false, b1, (short)0, acc01, false, false);
    acc10 = __builtin_amdgcn_wmma_f32_16x16x32_f16(false, a1, false, b0, (short)0, acc10, false, false);
    acc11 = __builtin_amdgcn_wmma_f32_16x16x32_f16(false, a1, false, b1, (short)0, acc11, false, false);

    if (kt + 1 < nk) s_store(1 - cur);
    __syncthreads();
  }

  // --- epilogue: C/D layout: lane -> n = l16; VGPR r -> m = r + 8*half
  v8f* accs[2][2] = { {&acc00, &acc01}, {&acc10, &acc11} };
#pragma unroll
  for (int i = 0; i < 2; ++i) {
#pragma unroll
    for (int j = 0; j < 2; ++j) {
      int n = nBlk + wn * 32 + j * 16 + l16;
#pragma unroll
      for (int r = 0; r < 8; ++r) {
        int m = mBlk + wm * 32 + i * 16 + r + 8 * half;
        if (m < M && n < N) {
          float v = (*accs[i][j])[r];
          if (scaleM) v *= scaleM[m];
          if (biasM)  v += biasM[m];
          if (biasN)  v += biasN[n];
          if (relu)   v = v > 0.f ? v : 0.f;
          C[(size_t)m * N + n] = v;
        }
      }
    }
  }
}

// ---------------------------------------------------------------------------
// Elementwise / data-movement kernels
// ---------------------------------------------------------------------------
__global__ void k_f32_to_f16(const float* __restrict__ in, h16* __restrict__ out, int n) {
  int i = blockIdx.x * blockDim.x + threadIdx.x;
  if (i < n) out[i] = (h16)in[i];
}

// in: [C, NX] f32 -> out: [NX, C] f16   (coalesced writes)
__global__ void k_transpose_cx_to_xc(const float* __restrict__ in, h16* __restrict__ out,
                                     int C, int NX) {
  int i = blockIdx.x * blockDim.x + threadIdx.x;
  if (i >= C * NX) return;
  int x = i / C;
  int c = i % C;
  out[i] = (h16)in[(size_t)c * NX + x];
}

// x: [C,H,W] f32, 3x3 SAME -> col: [H*W, C*9] f16  (K index = c*9 + kh*3 + kw)
__global__ void k_im2col3(const float* __restrict__ x, h16* __restrict__ col,
                          int C, int H, int W) {
  int idx = blockIdx.x * blockDim.x + threadIdx.x;
  if (idx >= H * W * C) return;
  int c  = idx % C;
  int hw = idx / C;
  int h = hw / W, w = hw % W;
  h16* o = col + (size_t)hw * (C * 9) + c * 9;
#pragma unroll
  for (int kh = 0; kh < 3; ++kh)
#pragma unroll
    for (int kw = 0; kw < 3; ++kw) {
      int hh = h + kh - 1, ww = w + kw - 1;
      float v = (hh >= 0 && hh < H && ww >= 0 && ww < W)
                ? x[((size_t)c * H + hh) * W + ww] : 0.f;
      o[kh * 3 + kw] = (h16)v;
    }
}

// dst[B,C,H,W] += nearest-up2(src[B,C,H/2,W/2])
__global__ void k_up2_add(float* __restrict__ dst, const float* __restrict__ src,
                          int B, int C, int H, int W) {
  size_t i = (size_t)blockIdx.x * blockDim.x + threadIdx.x;
  size_t total = (size_t)B * C * H * W;
  if (i >= total) return;
  int w = (int)(i % W); size_t t = i / W;
  int h = (int)(t % H); t /= H;
  int c = (int)(t % C); int b = (int)(t / C);
  dst[i] += src[(((size_t)b * C + c) * (H / 2) + h / 2) * (W / 2) + w / 2];
}

// fuse[B,256,128,128] = concat(up8(p5c), up4(p4c), up2(p3c), p2c)
__global__ void k_fuse_gather(float* __restrict__ fuse,
                              const float* __restrict__ p5, const float* __restrict__ p4,
                              const float* __restrict__ p3, const float* __restrict__ p2) {
  size_t i = (size_t)blockIdx.x * blockDim.x + threadIdx.x;
  size_t total = (size_t)4 * 256 * 128 * 128;
  if (i >= total) return;
  int x = (int)(i % 128); size_t t = i / 128;
  int y = (int)(t % 128); t /= 128;
  int ch = (int)(t % 256); int b = (int)(t / 256);
  float v;
  if (ch < 64)       v = p5[(((size_t)b * 64 + ch)        * 16 + y / 8) * 16 + x / 8];
  else if (ch < 128) v = p4[(((size_t)b * 64 + (ch - 64))  * 32 + y / 4) * 32 + x / 4];
  else if (ch < 192) v = p3[(((size_t)b * 64 + (ch - 128)) * 64 + y / 2) * 64 + x / 2];
  else               v = p2[(((size_t)b * 64 + (ch - 192)) * 128 + y)   * 128 + x];
  fuse[i] = v;
}

// ConvT 2x2 weight [C,O,2,2] -> A2[(tap*O+o), c] f16   (tap = kh*2+kw)
__global__ void k_repack_ct(const float* __restrict__ w, h16* __restrict__ A2, int C, int O) {
  int i = blockIdx.x * blockDim.x + threadIdx.x;
  if (i >= C * O * 4) return;
  int c = i % C; int t = i / C;
  int o = t % O; int tap = t / O;
  int kh = tap >> 1, kw = tap & 1;
  A2[i] = (h16)w[(((size_t)c * O + o) * 2 + kh) * 2 + kw];
}

__global__ void k_bn_scale(const float* __restrict__ w, float* __restrict__ s, int n) {
  int i = blockIdx.x * blockDim.x + threadIdx.x;
  if (i < n) s[i] = w[i] * rsqrtf(1.0f + 1e-5f);
}

// Cg [4*O, H*W] -> outT [(2H*2W), O] f16, with convT bias + BN(scale s,bias bb) + ReLU
__global__ void k_ct_scatter_hwc(const float* __restrict__ Cg, const float* __restrict__ ctb,
                                 const float* __restrict__ s, const float* __restrict__ bb,
                                 h16* __restrict__ outT, int O, int H, int W) {
  size_t i = (size_t)blockIdx.x * blockDim.x + threadIdx.x;
  size_t HW = (size_t)H * W;
  if (i >= 4 * (size_t)O * HW) return;
  int hw = (int)(i % HW); size_t t = i / HW;
  int o = (int)(t % O); int tap = (int)(t / O);
  int h = hw / W, w = hw % W;
  int kh = tap >> 1, kw = tap & 1;
  float v = (Cg[i] + ctb[o]) * s[o] + bb[o];
  v = v > 0.f ? v : 0.f;
  outT[((size_t)(2 * h + kh) * (2 * W) + (2 * w + kw)) * O + o] = (h16)v;
}

// Cg [4*O, H*W] -> out NCHW [O, 2H, 2W] f16, convT bias + BN + ReLU
__global__ void k_ct_scatter_nchw(const float* __restrict__ Cg, const float* __restrict__ ctb,
                                  const float* __restrict__ s, const float* __restrict__ bb,
                                  h16* __restrict__ out, int O, int H, int W) {
  size_t i = (size_t)blockIdx.x * blockDim.x + threadIdx.x;
  size_t HW = (size_t)H * W;
  if (i >= 4 * (size_t)O * HW) return;
  int hw = (int)(i % HW); size_t t = i / HW;
  int o = (int)(t % O); int tap = (int)(t / O);
  int h = hw / W, w = hw % W;
  int kh = tap >> 1, kw = tap & 1;
  float v = (Cg[i] + ctb[o]) * s[o] + bb[o];
  v = v > 0.f ? v : 0.f;
  out[((size_t)o * (2 * H) + (2 * h + kh)) * (2 * W) + (2 * w + kw)] = (h16)v;
}

// PositionEmbeddingSine, H=W=16, num_feats=128; pe [256][256]
__global__ void k_pos_sine(float* __restrict__ pe) {
  int i = blockIdx.x * blockDim.x + threadIdx.x;
  if (i >= 256 * 256) return;
  int hw = i % 256; int ch = i / 256;
  int h = hw / 16, w = hw % 16;
  const float scale = 6.283185307179586f;
  int f; float v;
  if (ch < 128) { f = ch;       v = (float)(h + 1) / (16.0f + 1e-6f) * scale; }
  else          { f = ch - 128; v = (float)(w + 1) / (16.0f + 1e-6f) * scale; }
  float dimt = powf(10000.0f, (2.0f * (float)(f / 2)) / 128.0f);
  float p = v / dimt;
  pe[i] = (f & 1) ? cosf(p) : sinf(p);
}

// gfC [B][256(d)][256(m)] + pe [256(d)][256(m)] -> mem/memp f16 rows r=b*256+m
__global__ void k_build_mem(const float* __restrict__ gfC, const float* __restrict__ pe,
                            h16* __restrict__ mem_h, h16* __restrict__ memp_h) {
  int i = blockIdx.x * blockDim.x + threadIdx.x;
  if (i >= 4 * 256 * 256) return;
  int d = i % 256; int t = i / 256;
  int m = t % 256; int b = t / 256;
  float mv = gfC[(((size_t)b * 256) + d) * 256 + m];
  float pv = pe[(size_t)d * 256 + m];
  mem_h[i]  = (h16)mv;
  memp_h[i] = (h16)(mv + pv);
}

// tgt=0, qpos rows r=l*4+b
__global__ void k_init_tgt(const float* __restrict__ qe, float* __restrict__ tgt,
                           float* __restrict__ qpos) {
  int i = blockIdx.x * blockDim.x + threadIdx.x;
  if (i >= 20 * 256) return;
  int d = i % 256; int r = i / 256; int l = r / 4;
  qpos[i] = qe[l * 256 + d];
  tgt[i] = 0.f;
}

__global__ void k_add_f16out(const float* __restrict__ a, const float* __restrict__ b,
                             h16* __restrict__ o, int n) {
  int i = blockIdx.x * blockDim.x + threadIdx.x;
  if (i < n) o[i] = (h16)(a[i] + b[i]);
}
__global__ void k_copy_f16(const float* __restrict__ a, h16* __restrict__ o, int n) {
  int i = blockIdx.x * blockDim.x + threadIdx.x;
  if (i < n) o[i] = (h16)a[i];
}
__global__ void k_gelu_f16(const float* __restrict__ in, h16* __restrict__ out, int n) {
  int i = blockIdx.x * blockDim.x + threadIdx.x;
  if (i >= n) return;
  float x = in[i];
  out[i] = (h16)(0.5f * x * (1.f + erff(x * 0.70710678118654752f)));
}

// tgt[r,:] = LayerNorm(tgt[r,:] + delta[r,:]) ; one 256-thread block per row
__global__ void k_residual_ln(float* __restrict__ tgt, const float* __restrict__ delta,
                              const float* __restrict__ w, const float* __restrict__ b2) {
  const int D = 256;
  int r = blockIdx.x, d = threadIdx.x;
  __shared__ float red[256];
  __shared__ float mean_s, var_s;
  float x = tgt[(size_t)r * D + d] + delta[(size_t)r * D + d];
  red[d] = x; __syncthreads();
  for (int s = 128; s > 0; s >>= 1) { if (d < s) red[d] += red[d + s]; __syncthreads(); }
  if (d == 0) mean_s = red[0] / D;
  __syncthreads();
  float c = x - mean_s;
  red[d] = c * c; __syncthreads();
  for (int s = 128; s > 0; s >>= 1) { if (d < s) red[d] += red[d + s]; __syncthreads(); }
  if (d == 0) var_s = red[0] / D;
  __syncthreads();
  tgt[(size_t)r * D + d] = c * rsqrtf(var_s + 1e-5f) * w[d] + b2[d];
}

// Attention: one thread per (b,h,l), online softmax; hd=32, NH=8
__global__ void k_attn(const float* __restrict__ Q, int ldq,
                       const float* __restrict__ Kp, int ldk,
                       const float* __restrict__ Vp, int ldv,
                       float* __restrict__ Op, int ldo,
                       int Lq, int Lk, int B, int kv_bmajor) {
  const int NHh = 8, hd = 32;
  int i = blockIdx.x * blockDim.x + threadIdx.x;
  if (i >= B * NHh * Lq) return;
  int l = i % Lq; int t = i / Lq; int h = t % NHh; int b = t / NHh;
  float qr[32];
  const float* q = Q + ((size_t)l * B + b) * ldq + h * hd;
  float scale = rsqrtf((float)hd);
#pragma unroll
  for (int j = 0; j < hd; ++j) qr[j] = q[j] * scale;
  float m_run = -1e30f, l_run = 0.f;
  float out[32];
#pragma unroll
  for (int j = 0; j < hd; ++j) out[j] = 0.f;
  for (int m = 0; m < Lk; ++m) {
    size_t rk = kv_bmajor ? ((size_t)b * Lk + m) : ((size_t)m * B + b);
    const float* k = Kp + rk * ldk + h * hd;
    const float* v = Vp + rk * ldv + h * hd;
    float s = 0.f;
#pragma unroll
    for (int j = 0; j < hd; ++j) s += qr[j] * k[j];
    float mn = fmaxf(m_run, s);
    float corr = __expf(m_run - mn);
    float p = __expf(s - mn);
    l_run = l_run * corr + p;
#pragma unroll
    for (int j = 0; j < hd; ++j) out[j] = out[j] * corr + p * v[j];
    m_run = mn;
  }
  float inv = 1.f / l_run;
  float* o = Op + ((size_t)l * B + b) * ldo + h * hd;
#pragma unroll
  for (int j = 0; j < hd; ++j) o[j] = out[j] * inv;
}

// kparams[b][q][c] = tgt[q*4+b,:] . kp_w[c,:] + kp_b[c]
__global__ void k_kparams(const float* __restrict__ tgt, const float* __restrict__ kpw,
                          const float* __restrict__ kpb, float* __restrict__ kp) {
  int i = blockIdx.x * blockDim.x + threadIdx.x;
  if (i >= 4 * 5 * 64) return;
  int c = i % 64; int t = i / 64;
  int q = t % 5; int b = t / 5;
  float s = kpb[c];
  const float* row = tgt + ((size_t)q * 4 + b) * 256;
  const float* wr = kpw + (size_t)c * 256;
  for (int d = 0; d < 256; ++d) s += row[d] * wr[d];
  kp[i] = s;
}

// masks: out[b,hw] = sigmoid(max_q sum_c fu[b,c,hw]*kp[b,q,c]) ; fu f16 NCHW
__global__ void k_final(const h16* __restrict__ fu, const float* __restrict__ kp,
                        float* __restrict__ out) {
  size_t i = (size_t)blockIdx.x * blockDim.x + threadIdx.x;
  const size_t HW = 512 * 512;
  if (i >= 4 * HW) return;
  size_t hw = i % HW; int b = (int)(i / HW);
  float x[64];
  const h16* base = fu + (size_t)b * 64 * HW + hw;
#pragma unroll
  for (int c = 0; c < 64; ++c) x[c] = (float)base[(size_t)c * HW];
  float best = -1e30f;
  for (int q = 0; q < 5; ++q) {
    const float* k = kp + ((size_t)b * 5 + q) * 64;
    float s = 0.f;
#pragma unroll
    for (int c = 0; c < 64; ++c) s += x[c] * k[c];
    best = fmaxf(best, s);
  }
  out[i] = 1.f / (1.f + __expf(-best));
}

// ---------------------------------------------------------------------------
// Host
// ---------------------------------------------------------------------------
static inline void gemm(hipStream_t st, const h16* A, const h16* B, float* C,
                        int M, int N, int K, const float* scaleM,
                        const float* biasM, const float* biasN, int relu) {
  dim3 g((N + TBN - 1) / TBN, (M + TBM - 1) / TBM);
  gemm_nt<<<g, 256, 0, st>>>(A, B, C, M, N, K, scaleM, biasM, biasN, relu);
}
#define EL(kern, n, ...) kern<<<((int)(((n) + 255) / 256)), 256, 0, stream>>>(__VA_ARGS__)

extern "C" void kernel_launch(void* const* d_in, const int* in_sizes, int n_in,
                              void* d_out, int out_size, void* d_ws, size_t ws_size,
                              hipStream_t stream) {
  (void)in_sizes; (void)n_in; (void)out_size; (void)ws_size;
  const float* c2   = (const float*)d_in[0];
  const float* c3   = (const float*)d_in[1];
  const float* c4   = (const float*)d_in[2];
  const float* c5   = (const float*)d_in[3];
  const float* in5w = (const float*)d_in[4];
  const float* in4w = (const float*)d_in[5];
  const float* in3w = (const float*)d_in[6];
  const float* in2w = (const float*)d_in[7];
  const float* o5w  = (const float*)d_in[8];
  const float* o4w  = (const float*)d_in[9];
  const float* o3w  = (const float*)d_in[10];
  const float* o2w  = (const float*)d_in[11];
  const float* gfrw = (const float*)d_in[12];
  const float* qemb = (const float*)d_in[13];
  const float* kpw  = (const float*)d_in[14];
  const float* kpb  = (const float*)d_in[15];
  const float* fuw  = (const float*)d_in[16];
  const float* bn1w = (const float*)d_in[17];
  const float* bn1b = (const float*)d_in[18];
  const float* ct1w = (const float*)d_in[19];
  const float* ct1b = (const float*)d_in[20];
  const float* bn2w = (const float*)d_in[21];
  const float* bn2b = (const float*)d_in[22];
  const float* ct2w = (const float*)d_in[23];
  const float* ct2b = (const float*)d_in[24];
  const float* bn3w = (const float*)d_in[25];
  const float* bn3b = (const float*)d_in[26];
  const float* saqkvw = (const float*)d_in[27];
  const float* saqkvb = (const float*)d_in[28];
  const float* saow   = (const float*)d_in[29];
  const float* saob   = (const float*)d_in[30];
  const float* caqkvw = (const float*)d_in[31];
  const float* caqkvb = (const float*)d_in[32];
  const float* caow   = (const float*)d_in[33];
  const float* caob   = (const float*)d_in[34];
  const float* ff1w   = (const float*)d_in[35];
  const float* ff1b   = (const float*)d_in[36];
  const float* ff2w   = (const float*)d_in[37];
  const float* ff2b   = (const float*)d_in[38];
  const float* ln1w   = (const float*)d_in[39];
  const float* ln1b   = (const float*)d_in[40];
  const float* ln2w   = (const float*)d_in[41];
  const float* ln2b   = (const float*)d_in[42];
  const float* ln3w   = (const float*)d_in[43];
  const float* ln3b   = (const float*)d_in[44];
  float* outp = (float*)d_out;

  // ---- workspace bump allocator (256B aligned) ----
  size_t off = 0;
  char* base = (char*)d_ws;
  auto alloc = [&](size_t bytes) -> void* {
    void* r = base + off;
    off += (bytes + 255) & ~(size_t)255;
    return r;
  };

  // f16 weights
  h16* w_in5 = (h16*)alloc(131072 * 2);
  h16* w_in4 = (h16*)alloc(65536 * 2);
  h16* w_in3 = (h16*)alloc(32768 * 2);
  h16* w_in2 = (h16*)alloc(16384 * 2);
  h16* w_o5  = (h16*)alloc(147456 * 2);
  h16* w_o4  = (h16*)alloc(147456 * 2);
  h16* w_o3  = (h16*)alloc(147456 * 2);
  h16* w_o2  = (h16*)alloc(147456 * 2);
  h16* w_fu  = (h16*)alloc(147456 * 2);
  h16* w_gfr = (h16*)alloc(131072 * 2);
  h16* w_ct1 = (h16*)alloc(16384 * 2);
  h16* w_ct2 = (h16*)alloc(16384 * 2);
  h16* w_saqkv = (h16*)alloc((size_t)1179648 * 2);
  h16* w_saout = (h16*)alloc((size_t)393216 * 2);
  h16* w_caqkv = (h16*)alloc((size_t)1179648 * 2);
  h16* w_caout = (h16*)alloc((size_t)393216 * 2);
  h16* w_ff1   = (h16*)alloc((size_t)3145728 * 2);
  h16* w_ff2   = (h16*)alloc((size_t)3145728 * 2);
  float* s1 = (float*)alloc(64 * 4);
  float* s2 = (float*)alloc(64 * 4);
  float* s3 = (float*)alloc(64 * 4);

  // activations
  float* t_in5 = (float*)alloc((size_t)4 * 256 * 256 * 4);
  float* t_in4 = (float*)alloc((size_t)4 * 256 * 1024 * 4);
  float* t_in3 = (float*)alloc((size_t)4 * 256 * 4096 * 4);
  float* t_in2 = (float*)alloc((size_t)4 * 256 * 16384 * 4);
  float* p5c = (float*)alloc((size_t)4 * 64 * 256 * 4);
  float* p4c = (float*)alloc((size_t)4 * 64 * 1024 * 4);
  float* p3c = (float*)alloc((size_t)4 * 64 * 4096 * 4);
  float* p2c = (float*)alloc((size_t)4 * 64 * 16384 * 4);
  float* t_fuse = (float*)alloc((size_t)4 * 256 * 16384 * 4);
  h16* xT  = (h16*)alloc((size_t)1048576 * 2);                    // 1x1 transposes
  h16* col = (h16*)alloc((size_t)16384 * 2304 * 2);               // im2col
  h16* x1T = (h16*)alloc((size_t)16384 * 64 * 2);
  h16* x2T = (h16*)alloc((size_t)65536 * 64 * 2);
  float* Cbig = (float*)alloc((size_t)256 * 65536 * 4);           // GEMM scratch
  h16* fuse_up = (h16*)alloc((size_t)4 * 64 * 262144 * 2);        // f16 NCHW
  float* gfC = (float*)alloc((size_t)4 * 256 * 256 * 4);
  float* pe  = (float*)alloc((size_t)256 * 256 * 4);
  h16* mem_h  = (h16*)alloc((size_t)1024 * 256 * 2);
  h16* memp_h = (h16*)alloc((size_t)1024 * 256 * 2);
  float* tgt  = (float*)alloc(20 * 256 * 4);
  float* qpos = (float*)alloc(20 * 256 * 4);
  h16* q_h    = (h16*)alloc(20 * 256 * 2);
  h16* t_h    = (h16*)alloc(20 * 256 * 2);
  float* qkC  = (float*)alloc(20 * 768 * 4);
  float* vC   = (float*)alloc(20 * 256 * 4);
  float* attnO = (float*)alloc(20 * 256 * 4);
  h16* ao_h   = (h16*)alloc(20 * 256 * 2);
  float* delta = (float*)alloc(20 * 256 * 4);
  float* qC   = (float*)alloc(20 * 256 * 4);
  float* kC   = (float*)alloc((size_t)1024 * 256 * 4);
  float* vC2  = (float*)alloc((size_t)1024 * 256 * 4);
  float* ffC  = (float*)alloc(20 * 2048 * 4);
  h16* ff_h   = (h16*)alloc(20 * 2048 * 2);
  float* kpar = (float*)alloc(4 * 5 * 64 * 4);

  // ---- 0. weight conversion ----
  EL(k_f32_to_f16, 131072, in5w, w_in5, 131072);
  EL(k_f32_to_f16, 65536,  in4w, w_in4, 65536);
  EL(k_f32_to_f16, 32768,  in3w, w_in3, 32768);
  EL(k_f32_to_f16, 16384,  in2w, w_in2, 16384);
  EL(k_f32_to_f16, 147456, o5w, w_o5, 147456);
  EL(k_f32_to_f16, 147456, o4w, w_o4, 147456);
  EL(k_f32_to_f16, 147456, o3w, w_o3, 147456);
  EL(k_f32_to_f16, 147456, o2w, w_o2, 147456);
  EL(k_f32_to_f16, 147456, fuw, w_fu, 147456);
  EL(k_f32_to_f16, 131072, gfrw, w_gfr, 131072);
  EL(k_f32_to_f16, 1179648, saqkvw, w_saqkv, 1179648);
  EL(k_f32_to_f16, 393216,  saow,   w_saout, 393216);
  EL(k_f32_to_f16, 1179648, caqkvw, w_caqkv, 1179648);
  EL(k_f32_to_f16, 393216,  caow,   w_caout, 393216);
  EL(k_f32_to_f16, 3145728, ff1w, w_ff1, 3145728);
  EL(k_f32_to_f16, 3145728, ff2w, w_ff2, 3145728);
  EL(k_repack_ct, 16384, ct1w, w_ct1, 64, 64);
  EL(k_repack_ct, 16384, ct2w, w_ct2, 64, 64);
  EL(k_bn_scale, 64, bn1w, s1, 64);
  EL(k_bn_scale, 64, bn2w, s2, 64);
  EL(k_bn_scale, 64, bn3w, s3, 64);

  // ---- 1. FPN lateral 1x1 convs (per batch GEMMs, WMMA) + gf ----
  for (int b = 0; b < 4; ++b) {
    EL(k_transpose_cx_to_xc, 512 * 256, c5 + (size_t)b * 512 * 256, xT, 512, 256);
    gemm(stream, w_in5, xT, t_in5 + (size_t)b * 256 * 256, 256, 256, 512, 0, 0, 0, 0);
    gemm(stream, w_gfr, xT, gfC + (size_t)b * 256 * 256, 256, 256, 512, 0, 0, 0, 0);
    EL(k_transpose_cx_to_xc, 256 * 1024, c4 + (size_t)b * 256 * 1024, xT, 256, 1024);
    gemm(stream, w_in4, xT, t_in4 + (size_t)b * 256 * 1024, 256, 1024, 256, 0, 0, 0, 0);
    EL(k_transpose_cx_to_xc, 128 * 4096, c3 + (size_t)b * 128 * 4096, xT, 128, 4096);
    gemm(stream, w_in3, xT, t_in3 + (size_t)b * 256 * 4096, 256, 4096, 128, 0, 0, 0, 0);
    EL(k_transpose_cx_to_xc, 64 * 16384, c2 + (size_t)b * 64 * 16384, xT, 64, 16384);
    gemm(stream, w_in2, xT, t_in2 + (size_t)b * 256 * 16384, 256, 16384, 64, 0, 0, 0, 0);
  }
  // top-down pathway (in-place adds)
  EL(k_up2_add, (size_t)4 * 256 * 32 * 32,   t_in4, t_in5, 4, 256, 32, 32);
  EL(k_up2_add, (size_t)4 * 256 * 64 * 64,   t_in3, t_in4, 4, 256, 64, 64);
  EL(k_up2_add, (size_t)4 * 256 * 128 * 128, t_in2, t_in3, 4, 256, 128, 128);

  // ---- 2. FPN output 3x3 convs (im2col + WMMA GEMM) ----
  for (int b = 0; b < 4; ++b) {
    EL(k_im2col3, 256 * 256,   t_in5 + (size_t)b * 256 * 256,   col, 256, 16, 16);
    gemm(stream, w_o5, col, p5c + (size_t)b * 64 * 256, 64, 256, 2304, 0, 0, 0, 0);
    EL(k_im2col3, 256 * 1024,  t_in4 + (size_t)b * 256 * 1024,  col, 256, 32, 32);
    gemm(stream, w_o4, col, p4c + (size_t)b * 64 * 1024, 64, 1024, 2304, 0, 0, 0, 0);
    EL(k_im2col3, 256 * 4096,  t_in3 + (size_t)b * 256 * 4096,  col, 256, 64, 64);
    gemm(stream, w_o3, col, p3c + (size_t)b * 64 * 4096, 64, 4096, 2304, 0, 0, 0, 0);
    EL(k_im2col3, 256 * 16384, t_in2 + (size_t)b * 256 * 16384, col, 256, 128, 128);
    gemm(stream, w_o2, col, p2c + (size_t)b * 64 * 16384, 64, 16384, 2304, 0, 0, 0, 0);
  }
  EL(k_fuse_gather, (size_t)4 * 256 * 128 * 128, t_fuse, p5c, p4c, p3c, p2c);

  // ---- 3. feature_upsample: fu conv + 2x ConvT(2,2) with BN+ReLU ----
  for (int b = 0; b < 4; ++b) {
    EL(k_im2col3, 256 * 16384, t_fuse + (size_t)b * 256 * 16384, col, 256, 128, 128);
    gemm(stream, w_fu, col, Cbig, 64, 16384, 2304, s1, bn1b, 0, 1);
    EL(k_transpose_cx_to_xc, 64 * 16384, Cbig, x1T, 64, 16384);
    gemm(stream, w_ct1, x1T, Cbig, 256, 16384, 64, 0, 0, 0, 0);
    EL(k_ct_scatter_hwc, (size_t)4 * 64 * 16384, Cbig, ct1b, s2, bn2b, x2T, 64, 128, 128);
    gemm(stream, w_ct2, x2T, Cbig, 256, 65536, 64, 0, 0, 0, 0);
    EL(k_ct_scatter_nchw, (size_t)4 * 64 * 65536, Cbig, ct2b, s3, bn3b,
       fuse_up + (size_t)b * 64 * 262144, 64, 256, 256);
  }

  // ---- 4. transformer decoder ----
  EL(k_pos_sine, 256 * 256, pe);
  EL(k_build_mem, 4 * 256 * 256, gfC, pe, mem_h, memp_h);
  EL(k_init_tgt, 20 * 256, qemb, tgt, qpos);
  for (int i = 0; i < 6; ++i) {
    const h16* Wsa = w_saqkv + (size_t)i * 768 * 256;
    const h16* Wca = w_caqkv + (size_t)i * 768 * 256;
    // self-attn: q=k=tgt+qpos, v=tgt
    EL(k_add_f16out, 20 * 256, tgt, qpos, q_h, 20 * 256);
    gemm(stream, q_h, Wsa, qkC, 20, 512, 256, 0, 0, saqkvb + i * 768, 0);
    EL(k_copy_f16, 20 * 256, tgt, t_h, 20 * 256);
    gemm(stream, t_h, Wsa + (size_t)512 * 256, vC, 20, 256, 256, 0, 0, saqkvb + i * 768 + 512, 0);
    EL(k_attn, 160, qkC, 512, qkC + 256, 512, vC, 256, attnO, 256, 5, 5, 4, 0);
    EL(k_copy_f16, 20 * 256, attnO, ao_h, 20 * 256);
    gemm(stream, ao_h, w_saout + (size_t)i * 65536, delta, 20, 256, 256, 0, 0, saob + i * 256, 0);
    k_residual_ln<<<20, 256, 0, stream>>>(tgt, delta, ln1w + i * 256, ln1b + i * 256);
    // cross-attn: q=tgt+qpos, k=memp, v=mem
    EL(k_add_f16out, 20 * 256, tgt, qpos, q_h, 20 * 256);
    gemm(stream, q_h, Wca, qC, 20, 256, 256, 0, 0, caqkvb + i * 768, 0);
    gemm(stream, memp_h, Wca + (size_t)256 * 256, kC, 1024, 256, 256, 0, 0, caqkvb + i * 768 + 256, 0);
    gemm(stream, mem_h,  Wca + (size_t)512 * 256, vC2, 1024, 256, 256, 0, 0, caqkvb + i * 768 + 512, 0);
    EL(k_attn, 160, qC, 256, kC, 256, vC2, 256, attnO, 256, 5, 256, 4, 1);
    EL(k_copy_f16, 20 * 256, attnO, ao_h, 20 * 256);
    gemm(stream, ao_h, w_caout + (size_t)i * 65536, delta, 20, 256, 256, 0, 0, caob + i * 256, 0);
    k_residual_ln<<<20, 256, 0, stream>>>(tgt, delta, ln2w + i * 256, ln2b + i * 256);
    // FFN
    EL(k_copy_f16, 20 * 256, tgt, t_h, 20 * 256);
    gemm(stream, t_h, w_ff1 + (size_t)i * 2048 * 256, ffC, 20, 2048, 256, 0, 0, ff1b + i * 2048, 0);
    EL(k_gelu_f16, 20 * 2048, ffC, ff_h, 20 * 2048);
    gemm(stream, ff_h, w_ff2 + (size_t)i * 256 * 2048, delta, 20, 256, 2048, 0, 0, ff2b + i * 256, 0);
    k_residual_ln<<<20, 256, 0, stream>>>(tgt, delta, ln3w + i * 256, ln3b + i * 256);
  }

  // ---- 5. kernel prediction + dynamic conv + max + sigmoid ----
  EL(k_kparams, 4 * 5 * 64, tgt, kpw, kpb, kpar);
  EL(k_final, (size_t)4 * 512 * 512, fuse_up, kpar, outp);
}
